// OCN_57354993271385
// MI455X (gfx1250) — matrix operand order repaired and checked by reading, not compiled, not run on hardware
//
#include <hip/hip_runtime.h>
#include <hip/hip_bf16.h>

// ---------------------------------------------------------------------------
// OCN forward for MI455X (gfx1250, wave32). All heavy contractions run on
// V_WMMA_F32_16X16X32_BF16 (f32 accumulate, bf16 operands). ~0.4 TFLOP of
// GEMM vs ~134MB input -> matrix-unit bound; bf16 WMMA is the fast path.
// GEMM is register-blocked 32x32 per wave (2x2 WMMA tiles): 4 WMMAs per
// K-step against 4 fragment loads -> ~1KB/wave moved per WMMA (vs 2KB for a
// naive 16x16 tile), keeping the matrix pipe fed from L2/WGP$.
// Workspace requirement: ~600 MB (see carve-out below).
// doc_final_pos=768 / num_label=4 are device scalars in d_in; they are
// compile-time constants here (graph capture forbids d2h reads).
// ---------------------------------------------------------------------------

typedef __bf16 bf16;
typedef __attribute__((ext_vector_type(8)))  bf16  v8bf;
typedef __attribute__((ext_vector_type(16))) bf16  v16bf;
typedef __attribute__((ext_vector_type(8)))  float v8f;

#define EPI_NONE       0
#define EPI_ROWCOL     1   // += rowAdd[m] + colAdd[n]   (trilinear ql/kl bias)
#define EPI_BIAS       2   // += bias[n]
#define EPI_BIAS_TANH  3   // tanh(x + bias[n])
#define EPI_BIAS_RELU  4   // relu(x + bias[n])

__device__ __forceinline__ v16bf pack16(v8bf lo, v8bf hi) {
  v16bf r;
#pragma unroll
  for (int i = 0; i < 8; ++i) { r[i] = lo[i]; r[i + 8] = hi[i]; }
  return r;
}

// Batched GEMM, C[m,n] = sum_k A[m,k] * (BT ? B[n,k] : B[k,n]).
// Block = 128 threads = 4 waves arranged 2x2 -> 64x64 block tile.
// Each wave owns a 32x32 tile = 2x2 WMMA accumulators; K stepped by 32.
// Requires M%64==0, N%64==0, K%32==0 (true for every call site here).
// A/B fragment layouts follow CDNA5 ISA 7.12.2 (16-bit A 16x32; B 32x16).
template<bool BT, int EPI>
__global__ __launch_bounds__(128) void gemm_bf16_wmma(
    const bf16* __restrict__ A, long lda, long sA,
    const bf16* __restrict__ B, long ldb, long sB,
    float* __restrict__ C, long ldc, long sC,
    bf16*  __restrict__ Cb, long ldcb, long sCb,
    const float* __restrict__ rowAdd, long sRow,
    const float* __restrict__ colAdd, long sCol,
    const float* __restrict__ bias, int K)
{
  const int lane = threadIdx.x & 31;
  const int wave = threadIdx.x >> 5;
  const int m0 = (blockIdx.x << 6) + ((wave >> 1) << 5);
  const int n0 = (blockIdx.y << 6) + ((wave & 1) << 5);
  const long z = blockIdx.z;
  A += z * sA; B += z * sB;

  // A: lane holds row m+(lane&15); K pairs at +8*(lane>>4) and +16+8*(lane>>4)
  const bf16* ap0 = A + (long)(m0 + (lane & 15)) * lda + ((lane >> 4) << 3);
  const bf16* ap1 = ap0 + 16 * lda;

  const v8f vzero = {0.f, 0.f, 0.f, 0.f, 0.f, 0.f, 0.f, 0.f};
  v8f acc[2][2];
#pragma unroll
  for (int mi = 0; mi < 2; ++mi)
#pragma unroll
    for (int ni = 0; ni < 2; ++ni) acc[mi][ni] = vzero;

  if (BT) {
    // B stored (N,K) row-major: lane holds col n+(lane&15),
    // K runs of 8 at +8*(lane>>4) and +16+8*(lane>>4) -> two b128 loads per fragment.
    const bf16* bp0 = B + (long)(n0 + (lane & 15)) * ldb + ((lane >> 4) << 3);
    const bf16* bp1 = bp0 + 16 * ldb;
    for (int k0 = 0; k0 < K; k0 += 32) {
      v16bf a0 = pack16(*(const v8bf*)(ap0 + k0), *(const v8bf*)(ap0 + k0 + 16));
      v16bf a1 = pack16(*(const v8bf*)(ap1 + k0), *(const v8bf*)(ap1 + k0 + 16));
      v16bf b0 = pack16(*(const v8bf*)(bp0 + k0), *(const v8bf*)(bp0 + k0 + 16));
      v16bf b1 = pack16(*(const v8bf*)(bp1 + k0), *(const v8bf*)(bp1 + k0 + 16));
      acc[0][0] = __builtin_amdgcn_wmma_f32_16x16x32_bf16(false, a0, false, b0, (short)0, acc[0][0], false, false);
      acc[0][1] = __builtin_amdgcn_wmma_f32_16x16x32_bf16(false, a0, false, b1, (short)0, acc[0][1], false, false);
      acc[1][0] = __builtin_amdgcn_wmma_f32_16x16x32_bf16(false, a1, false, b0, (short)0, acc[1][0], false, false);
      acc[1][1] = __builtin_amdgcn_wmma_f32_16x16x32_bf16(false, a1, false, b1, (short)0, acc[1][1], false, false);
    }
  } else {
    // B stored (K,N) row-major: strided per-element gather for the fragments.
    const int bn0 = n0 + (lane & 15);
    const int bk = (lane >> 4) << 3;
    for (int k0 = 0; k0 < K; k0 += 32) {
      v16bf a0 = pack16(*(const v8bf*)(ap0 + k0), *(const v8bf*)(ap0 + k0 + 16));
      v16bf a1 = pack16(*(const v8bf*)(ap1 + k0), *(const v8bf*)(ap1 + k0 + 16));
      v16bf b0, b1;
#pragma unroll
      for (int i = 0; i < 16; ++i) {
        const bf16* bp = B + (long)(k0 + bk + (i < 8 ? i : i + 8)) * ldb + bn0;
        b0[i] = bp[0];
        b1[i] = bp[16];
      }
      acc[0][0] = __builtin_amdgcn_wmma_f32_16x16x32_bf16(false, a0, false, b0, (short)0, acc[0][0], false, false);
      acc[0][1] = __builtin_amdgcn_wmma_f32_16x16x32_bf16(false, a0, false, b1, (short)0, acc[0][1], false, false);
      acc[1][0] = __builtin_amdgcn_wmma_f32_16x16x32_bf16(false, a1, false, b0, (short)0, acc[1][0], false, false);
      acc[1][1] = __builtin_amdgcn_wmma_f32_16x16x32_bf16(false, a1, false, b1, (short)0, acc[1][1], false, false);
    }
  }

  // D layout: lane n = lane&15, VGPR r -> row + r + 8*(lane>>4)
  const int mh = (lane >> 4) << 3;
#pragma unroll
  for (int mi = 0; mi < 2; ++mi) {
#pragma unroll
    for (int ni = 0; ni < 2; ++ni) {
      const int cn = n0 + (ni << 4) + (lane & 15);
#pragma unroll
      for (int r = 0; r < 8; ++r) {
        const int row = m0 + (mi << 4) + mh + r;
        float v = acc[mi][ni][r];
        if (EPI == EPI_ROWCOL) v += rowAdd[z * sRow + row] + colAdd[z * sCol + cn];
        if (EPI == EPI_BIAS || EPI == EPI_BIAS_TANH || EPI == EPI_BIAS_RELU)
          v += bias[cn];
        if (EPI == EPI_BIAS_TANH) v = tanhf(v);
        if (EPI == EPI_BIAS_RELU) v = fmaxf(v, 0.f);
        if (C)  C [z * sC  + (long)row * ldc  + cn] = v;
        if (Cb) Cb[z * sCb + (long)row * ldcb + cn] = (bf16)v;
      }
    }
  }
}

// f32 -> bf16 strided copy. grid = (cols/256, rows, batch)
__global__ __launch_bounds__(256) void cvt_rows(
    const float* __restrict__ src, long lds, long sS,
    bf16* __restrict__ dst, long ldd, long sD)
{
  const int  c = blockIdx.x * 256 + threadIdx.x;
  const long r = blockIdx.y, z = blockIdx.z;
  dst[z * sD + r * ldd + c] = (bf16)src[z * sS + r * lds + c];
}

// Y[row,d] = X[row,d] * w[d]; contiguous rows of 1024, grid.z = batch.
__global__ __launch_bounds__(256) void scale_rows_bf16(
    const float* __restrict__ X, long sX, const float* __restrict__ w,
    bf16* __restrict__ Y, long sY)
{
  const long idx = (long)blockIdx.x * 256 + threadIdx.x;
  const long z = blockIdx.z;
  Y[z * sY + idx] = (bf16)(X[z * sX + idx] * w[idx & 1023]);
}

// out[row] = dot(X[row,:], w). One wave per row; 8 waves per block.
template<typename T>
__global__ __launch_bounds__(256) void rowdot(
    const T* __restrict__ X, long ld, long sX, const float* __restrict__ w,
    float* __restrict__ out, long sOut, int cols)
{
  const int lane = threadIdx.x & 31, wv = threadIdx.x >> 5;
  const long row = (long)blockIdx.x * 8 + wv, z = blockIdx.z;
  const T* p = X + z * sX + row * ld;
  float s = 0.f;
  for (int c = lane; c < cols; c += 32) s += (float)p[c] * w[c];
#pragma unroll
  for (int o = 16; o > 0; o >>= 1) s += __shfl_xor(s, o, 32);
  if (lane == 0) out[z * sOut + row] = s;
}

// Masked softmax over the last axis; writes bf16 attention weights.
__global__ __launch_bounds__(256) void softmax_rows(
    const float* __restrict__ L, long sL,
    const int* __restrict__ qm, long sQm, const int* __restrict__ km, long sKm,
    bf16* __restrict__ aw, long sAw, int Lk)
{
  const int lane = threadIdx.x & 31, wv = threadIdx.x >> 5;
  const long q = (long)blockIdx.x * 8 + wv, z = blockIdx.z;
  const float* row = L + z * sL + q * Lk;
  const int* kmp = km + z * sKm;
  const int qv = qm[z * sQm + q];
  float mx = -__builtin_inff();
  for (int k = lane; k < Lk; k += 32) if (kmp[k]) mx = fmaxf(mx, row[k]);
#pragma unroll
  for (int o = 16; o > 0; o >>= 1) mx = fmaxf(mx, __shfl_xor(mx, o, 32));
  float sum = 0.f;
  for (int k = lane; k < Lk; k += 32) if (kmp[k]) sum += __expf(row[k] - mx);
#pragma unroll
  for (int o = 16; o > 0; o >>= 1) sum += __shfl_xor(sum, o, 32);
  const float inv = (qv && sum > 0.f) ? 1.f / sum : 0.f;
  for (int k = lane; k < Lk; k += 32) {
    const float v = (qv && kmp[k]) ? __expf(row[k] - mx) * inv : 0.f;
    aw[z * sAw + q * Lk + k] = (bf16)v;
  }
}

// Masked softmax over axis=1 (columns): wave per column k.
__global__ __launch_bounds__(256) void softmax_cols(
    const float* __restrict__ L, long sL,
    const int* __restrict__ qm, long sQm, const int* __restrict__ km, long sKm,
    bf16* __restrict__ kq, long sKq, int Lq, int Lk)
{
  const int lane = threadIdx.x & 31, wv = threadIdx.x >> 5;
  const long k = (long)blockIdx.x * 8 + wv, z = blockIdx.z;
  const float* base = L + z * sL;
  const int* qmp = qm + z * sQm;
  const int kv = km[z * sKm + k];
  float mx = -__builtin_inff();
  for (int q = lane; q < Lq; q += 32) if (qmp[q]) mx = fmaxf(mx, base[(long)q * Lk + k]);
#pragma unroll
  for (int o = 16; o > 0; o >>= 1) mx = fmaxf(mx, __shfl_xor(mx, o, 32));
  float sum = 0.f;
  for (int q = lane; q < Lq; q += 32) if (qmp[q]) sum += __expf(base[(long)q * Lk + k] - mx);
#pragma unroll
  for (int o = 16; o > 0; o >>= 1) sum += __shfl_xor(sum, o, 32);
  const float inv = (kv && sum > 0.f) ? 1.f / sum : 0.f;
  for (int q = lane; q < Lq; q += 32) {
    const float v = (kv && qmp[q]) ? __expf(base[(long)q * Lk + k] - mx) * inv : 0.f;
    kq[z * sKq + (long)q * Lk + k] = (bf16)v;
  }
}

// comp_in row = [cur | cur*a0 | cur-a0 | cur*a1 | cur-a1 | cur*a2 | cur-a2]
__global__ __launch_bounds__(256) void build_comp_in(
    const float* __restrict__ cur, long sCur,
    const float* __restrict__ a0, const float* __restrict__ a1,
    const float* __restrict__ a2, bf16* __restrict__ out)
{
  const int d = blockIdx.x * 256 + threadIdx.x, q = blockIdx.y;
  const long b = blockIdx.z;
  const long off = (long)q * 1024 + d;
  const float c = cur[b * sCur + off];
  const float x0 = a0[b * 262144 + off];
  const float x1 = a1[b * 262144 + off];
  const float x2 = a2[b * 262144 + off];
  bf16* o = out + b * 1835008 + (long)q * 7168;
  o[d]        = (bf16)c;
  o[1024 + d] = (bf16)(c * x0);  o[2048 + d] = (bf16)(c - x0);
  o[3072 + d] = (bf16)(c * x1);  o[4096 + d] = (bf16)(c - x1);
  o[5120 + d] = (bf16)(c * x2);  o[6144 + d] = (bf16)(c - x2);
}

// option = oe*sig(g) + oc*(1-sig(g)); bf16 into attn_in slot0 (ld 3072)
__global__ __launch_bounds__(256) void gate_combine(
    const float* __restrict__ oe, long sOe, const float* __restrict__ oc,
    const float* __restrict__ g, bf16* __restrict__ out, long sOut)
{
  const long idx = (long)blockIdx.x * 256 + threadIdx.x, z = blockIdx.z;
  const int q = (int)(idx >> 10), d = (int)(idx & 1023);
  const float gv = 1.f / (1.f + __expf(-g[z * 262144 + idx]));
  const float v = oe[z * sOe + idx] * gv + oc[z * 262144 + idx] * (1.f - gv);
  out[z * sOut + (long)q * 3072 + d] = (bf16)v;
}

// self_in slots 2,3 = fusion*sa, fusion-sa (slots 0,1 written by GEMM Cbf)
__global__ __launch_bounds__(256) void build_self_in(
    const float* __restrict__ fu, const float* __restrict__ sa,
    bf16* __restrict__ out)
{
  const long idx = (long)blockIdx.x * 256 + threadIdx.x, z = blockIdx.z;
  const int q = (int)(idx >> 10), d = (int)(idx & 1023);
  const float f = fu[z * 262144 + idx], s = sa[z * 262144 + idx];
  bf16* o = out + z * 1048576 + (long)q * 4096;
  o[2048 + d] = (bf16)(f * s);
  o[3072 + d] = (bf16)(f - s);
}

// out[s,h] = max over masked option rows of X[s,q,h] (else -inf)
__global__ __launch_bounds__(256) void masked_colmax(
    const float* __restrict__ X, const int* __restrict__ am,
    float* __restrict__ out)
{
  const int h = blockIdx.x * 256 + threadIdx.x;
  const long s = blockIdx.y;
  const int* m = am + s * 1024 + 768;
  const float* x = X + s * 262144;
  float mx = -__builtin_inff();
  for (int q = 0; q < 256; ++q) if (m[q]) mx = fmaxf(mx, x[(long)q * 1024 + h]);
  out[s * 1024 + h] = mx;
}

template<bool BT, int EPI>
static inline void gemm(hipStream_t st, int M, int N, int K, int batch,
    const bf16* A, long lda, long sA, const bf16* B, long ldb, long sB,
    float* C, long ldc, long sC, bf16* Cb, long ldcb, long sCb,
    const float* rA, long srA, const float* cA, long scA, const float* bias)
{
  gemm_bf16_wmma<BT, EPI><<<dim3(M / 64, N / 64, batch), dim3(128), 0, st>>>(
      A, lda, sA, B, ldb, sB, C, ldc, sC, Cb, ldcb, sCb, rA, srA, cA, scA, bias, K);
}

extern "C" void kernel_launch(void* const* d_in, const int* in_sizes, int n_in,
                              void* d_out, int out_size, void* d_ws, size_t ws_size,
                              hipStream_t stream)
{
  (void)in_sizes; (void)n_in; (void)out_size; (void)ws_size;
  constexpr int SEQ = 1024, Hc = 1024, Lo = 256, Ld = 768, NSEQ = 32, NB = 8, NL = 4;

  const int*   am     = (const int*)  d_in[1];
  const float* LL     = (const float*)d_in[4];
  const float* ow1 = (const float*)d_in[5],  *ow2 = (const float*)d_in[6],  *ow3 = (const float*)d_in[7];
  const float* dw1 = (const float*)d_in[8],  *dw2 = (const float*)d_in[9],  *dw3 = (const float*)d_in[10];
  const float* sw1 = (const float*)d_in[11], *sw2 = (const float*)d_in[12], *sw3 = (const float*)d_in[13];
  const float* comp_w = (const float*)d_in[14], *comp_b = (const float*)d_in[15];
  const float* gate_w = (const float*)d_in[16], *gate_b = (const float*)d_in[17];
  const float* attn_w = (const float*)d_in[18], *attn_b = (const float*)d_in[19];
  const float* self_w = (const float*)d_in[20], *self_b = (const float*)d_in[21];
  float* out = (float*)d_out;

  // ---- workspace carve-out (total ~596 MB) ----
  char* p = (char*)d_ws;
  auto alloc = [&](size_t bytes) -> void* {
    void* r = (void*)p; p += (bytes + 255) & ~(size_t)255; return r;
  };
  bf16*  bfL      = (bf16*) alloc((size_t)NSEQ * SEQ * Hc * 2);   // bf16 last_layer
  bf16*  comp_wb  = (bf16*) alloc((size_t)Hc * 7 * Hc * 2);
  bf16*  gate_wb  = (bf16*) alloc((size_t)Hc * 2 * Hc * 2);
  bf16*  attn_wb  = (bf16*) alloc((size_t)Hc * 3 * Hc * 2);
  bf16*  self_wb  = (bf16*) alloc((size_t)Hc * 4 * Hc * 2);
  bf16*  optK     = (bf16*) alloc((size_t)NSEQ * Lo * Hc * 2);    // also reused as selfK
  bf16*  docK     = (bf16*) alloc((size_t)NSEQ * Ld * Hc * 2);
  float* qlo      = (float*)alloc((size_t)NSEQ * Lo * 4);
  float* klo      = (float*)alloc((size_t)NSEQ * Lo * 4);
  float* qld      = (float*)alloc((size_t)NSEQ * Lo * 4);
  float* kld      = (float*)alloc((size_t)NSEQ * Ld * 4);
  float* qls      = (float*)alloc((size_t)NSEQ * Lo * 4);
  float* kls      = (float*)alloc((size_t)NSEQ * Lo * 4);
  float* logits   = (float*)alloc((size_t)NSEQ * Lo * Ld * 4);    // covers all logit shapes
  bf16*  aw       = (bf16*) alloc((size_t)NSEQ * Lo * Ld * 2);
  bf16*  kq       = (bf16*) alloc((size_t)NSEQ * Lo * Ld * 2);
  float* abuf     = (float*)alloc((size_t)3 * NB * Lo * Hc * 4);  // a for 3 opposing options
  bf16*  comp_in  = (bf16*) alloc((size_t)NB * Lo * 7 * Hc * 2);
  float* opt_corr = (float*)alloc((size_t)NSEQ * Lo * Hc * 4);
  bf16*  gate_in  = (bf16*) alloc((size_t)NSEQ * Lo * 2 * Hc * 2);
  float* g_lin    = (float*)alloc((size_t)NSEQ * Lo * Hc * 4);
  bf16*  attn_in  = (bf16*) alloc((size_t)NSEQ * Lo * 3 * Hc * 2);
  bf16*  co_wb    = (bf16*) alloc((size_t)NSEQ * Lo * Lo * 2);
  float* fusion   = (float*)alloc((size_t)NSEQ * Lo * Hc * 4);
  bf16*  self_in  = (bf16*) alloc((size_t)NSEQ * Lo * 4 * Hc * 2);
  float* sa       = (float*)alloc((size_t)NSEQ * Lo * Hc * 4);
  float* fusion2  = (float*)alloc((size_t)NSEQ * Lo * Hc * 4);

  const long seqStr = (long)SEQ * Hc;      // 1048576
  const long loH    = (long)Lo * Hc;       // 262144

  // ---- bf16 conversions (activations + weights) ----
  cvt_rows<<<dim3(Hc / 256, SEQ, NSEQ), 256, 0, stream>>>(LL, Hc, seqStr, bfL, Hc, seqStr);
  cvt_rows<<<dim3(7 * Hc / 256, Hc, 1), 256, 0, stream>>>(comp_w, 7 * Hc, 0, comp_wb, 7 * Hc, 0);
  cvt_rows<<<dim3(2 * Hc / 256, Hc, 1), 256, 0, stream>>>(gate_w, 2 * Hc, 0, gate_wb, 2 * Hc, 0);
  cvt_rows<<<dim3(3 * Hc / 256, Hc, 1), 256, 0, stream>>>(attn_w, 3 * Hc, 0, attn_wb, 3 * Hc, 0);
  cvt_rows<<<dim3(4 * Hc / 256, Hc, 1), 256, 0, stream>>>(self_w, 4 * Hc, 0, self_wb, 4 * Hc, 0);

  // ---- stage A: option-vs-option attention + comp projection ----
  scale_rows_bf16<<<dim3(loH / 256, 1, NSEQ), 256, 0, stream>>>(LL + 768 * Hc, seqStr, ow3, optK, loH);
  rowdot<float><<<dim3(Lo / 8, 1, NSEQ), 256, 0, stream>>>(LL + 768 * Hc, Hc, seqStr, ow1, qlo, Lo, Hc);
  rowdot<float><<<dim3(Lo / 8, 1, NSEQ), 256, 0, stream>>>(LL + 768 * Hc, Hc, seqStr, ow2, klo, Lo, Hc);

  for (int i = 0; i < NL; ++i) {
    int t = 0;
    for (int j = 0; j < NL; ++j) {
      if (j == i) continue;
      // logits[q,k] = cur_q . (opt_j,k * ow3) + ql[q] + kl[k]
      gemm<true, EPI_ROWCOL>(stream, Lo, Lo, Hc, NB,
          bfL + ((long)i * SEQ + 768) * Hc, Hc, (long)NL * seqStr,
          optK + (long)j * loH, Hc, (long)NL * loH,
          logits, Lo, (long)Lo * Lo, nullptr, 0, 0,
          qlo + i * Lo, (long)NL * Lo, klo + j * Lo, (long)NL * Lo, nullptr);
      softmax_rows<<<dim3(Lo / 8, 1, NB), 256, 0, stream>>>(logits, (long)Lo * Lo,
          am + i * SEQ + 768, (long)NL * SEQ, am + j * SEQ + 768, (long)NL * SEQ,
          aw, (long)Lo * Lo, Lo);
      // a = aw @ opt_j   (NN: B = V rows)
      gemm<false, EPI_NONE>(stream, Lo, Hc, Lo, NB,
          aw, Lo, (long)Lo * Lo,
          bfL + ((long)j * SEQ + 768) * Hc, Hc, (long)NL * seqStr,
          abuf + (long)t * NB * loH, Hc, loH, nullptr, 0, 0,
          nullptr, 0, nullptr, 0, nullptr);
      ++t;
    }
    build_comp_in<<<dim3(Hc / 256, Lo, NB), 256, 0, stream>>>(
        LL + ((long)i * SEQ + 768) * Hc, (long)NL * seqStr,
        abuf, abuf + (long)NB * loH, abuf + (long)2 * NB * loH, comp_in);
    // opt_corr = tanh(comp_in @ comp_w^T + b); bf16 copy straight into gate_in slot1
    gemm<true, EPI_BIAS_TANH>(stream, Lo, Hc, 7 * Hc, NB,
        comp_in, 7 * Hc, (long)Lo * 7 * Hc,
        comp_wb, 7 * Hc, 0,
        opt_corr + (long)i * loH, Hc, (long)NL * loH,
        gate_in + (long)i * Lo * 2 * Hc + Hc, 2 * Hc, (long)NL * Lo * 2 * Hc,
        nullptr, 0, nullptr, 0, comp_b);
  }

  // ---- gate + combine ----
  cvt_rows<<<dim3(Hc / 256, Lo, NSEQ), 256, 0, stream>>>(LL + 768 * Hc, Hc, seqStr,
                                                         gate_in, 2 * Hc, (long)Lo * 2 * Hc);
  gemm<true, EPI_BIAS>(stream, Lo, Hc, 2 * Hc, NSEQ,
      gate_in, 2 * Hc, (long)Lo * 2 * Hc, gate_wb, 2 * Hc, 0,
      g_lin, Hc, loH, nullptr, 0, 0, nullptr, 0, nullptr, 0, gate_b);
  gate_combine<<<dim3(loH / 256, 1, NSEQ), 256, 0, stream>>>(
      LL + 768 * Hc, seqStr, opt_corr, g_lin, attn_in, (long)Lo * 3 * Hc);

  // ---- doc attention (attn + co_attn) ----
  scale_rows_bf16<<<dim3((long)Ld * Hc / 256, 1, NSEQ), 256, 0, stream>>>(LL, seqStr, dw3, docK, (long)Ld * Hc);
  rowdot<bf16> <<<dim3(Lo / 8, 1, NSEQ), 256, 0, stream>>>(attn_in, 3 * Hc, (long)Lo * 3 * Hc, dw1, qld, Lo, Hc);
  rowdot<float><<<dim3(Ld / 8, 1, NSEQ), 256, 0, stream>>>(LL, Hc, seqStr, dw2, kld, Ld, Hc);
  gemm<true, EPI_ROWCOL>(stream, Lo, Ld, Hc, NSEQ,
      attn_in, 3 * Hc, (long)Lo * 3 * Hc, docK, Hc, (long)Ld * Hc,
      logits, Ld, (long)Lo * Ld, nullptr, 0, 0,
      qld, Lo, kld, Ld, nullptr);
  softmax_rows<<<dim3(Lo / 8, 1, NSEQ), 256, 0, stream>>>(logits, (long)Lo * Ld,
      am + 768, SEQ, am, SEQ, aw, (long)Lo * Ld, Ld);
  softmax_cols<<<dim3(Ld / 8, 1, NSEQ), 256, 0, stream>>>(logits, (long)Lo * Ld,
      am + 768, SEQ, am, SEQ, kq, (long)Lo * Ld, Lo, Ld);
  // a = aw @ doc  -> attn_in slot1 (bf16 only)
  gemm<false, EPI_NONE>(stream, Lo, Hc, Ld, NSEQ,
      aw, Ld, (long)Lo * Ld, bfL, Hc, seqStr,
      nullptr, 0, 0, attn_in + Hc, 3 * Hc, (long)Lo * 3 * Hc,
      nullptr, 0, nullptr, 0, nullptr);
  // co_w = aw @ kq^T  (NT with B = kq rows)
  gemm<true, EPI_NONE>(stream, Lo, Lo, Ld, NSEQ,
      aw, Ld, (long)Lo * Ld, kq, Ld, (long)Lo * Ld,
      nullptr, 0, 0, co_wb, Lo, (long)Lo * Lo,
      nullptr, 0, nullptr, 0, nullptr);
  // co_attn = co_w @ option -> attn_in slot2
  gemm<false, EPI_NONE>(stream, Lo, Hc, Lo, NSEQ,
      co_wb, Lo, (long)Lo * Lo, attn_in, 3 * Hc, (long)Lo * 3 * Hc,
      nullptr, 0, 0, attn_in + 2 * Hc, 3 * Hc, (long)Lo * 3 * Hc,
      nullptr, 0, nullptr, 0, nullptr);
  // fusion = relu([option|a|co] @ attn_w^T + b); bf16 into self_in slot0
  gemm<true, EPI_BIAS_RELU>(stream, Lo, Hc, 3 * Hc, NSEQ,
      attn_in, 3 * Hc, (long)Lo * 3 * Hc, attn_wb, 3 * Hc, 0,
      fusion, Hc, loH, self_in, 4 * Hc, (long)Lo * 4 * Hc,
      nullptr, 0, nullptr, 0, attn_b);

  // ---- self attention ----
  scale_rows_bf16<<<dim3(loH / 256, 1, NSEQ), 256, 0, stream>>>(fusion, loH, sw3, optK, loH); // reuse as selfK
  rowdot<float><<<dim3(Lo / 8, 1, NSEQ), 256, 0, stream>>>(fusion, Hc, loH, sw1, qls, Lo, Hc);
  rowdot<float><<<dim3(Lo / 8, 1, NSEQ), 256, 0, stream>>>(fusion, Hc, loH, sw2, kls, Lo, Hc);
  gemm<true, EPI_ROWCOL>(stream, Lo, Lo, Hc, NSEQ,
      self_in, 4 * Hc, (long)Lo * 4 * Hc, optK, Hc, loH,
      logits, Lo, (long)Lo * Lo, nullptr, 0, 0,
      qls, Lo, kls, Lo, nullptr);
  softmax_rows<<<dim3(Lo / 8, 1, NSEQ), 256, 0, stream>>>(logits, (long)Lo * Lo,
      am + 768, SEQ, am + 768, SEQ, aw, (long)Lo * Lo, Lo);
  // sa = aw @ fusion (f32 for elementwise, bf16 into self_in slot1)
  gemm<false, EPI_NONE>(stream, Lo, Hc, Lo, NSEQ,
      aw, Lo, (long)Lo * Lo, self_in, 4 * Hc, (long)Lo * 4 * Hc,
      sa, Hc, loH, self_in + Hc, 4 * Hc, (long)Lo * 4 * Hc,
      nullptr, 0, nullptr, 0, nullptr);
  build_self_in<<<dim3(loH / 256, 1, NSEQ), 256, 0, stream>>>(fusion, sa, self_in);
  gemm<true, EPI_BIAS_RELU>(stream, Lo, Hc, 4 * Hc, NSEQ,
      self_in, 4 * Hc, (long)Lo * 4 * Hc, self_wb, 4 * Hc, 0,
      fusion2, Hc, loH, nullptr, 0, 0,
      nullptr, 0, nullptr, 0, self_b);

  // ---- masked max over option positions -> (32, 1024) ----
  masked_colmax<<<dim3(Hc / 256, NSEQ), 256, 0, stream>>>(fusion2, am, out);
}
